// GCNCriticNet_82188494176621
// MI455X (gfx1250) — matrix operand order repaired and checked by reading, not compiled, not run on hardware
//
#include <hip/hip_runtime.h>

typedef __attribute__((ext_vector_type(16))) _Float16 v16h;
typedef __attribute__((ext_vector_type(8)))  float    v8f;

#define NAGENTS 16
#define NBATCH  8192
#define OBSD    64
#define HIDD    128
#define WAVES_PER_BLK 8
#define BLK_THREADS   256

// Packed B-fragment pool in d_ws (f16):
//   fragment block b = {emb: kc*8+nt (0..15)} {gcn layer l: 16 + 32*l + kc*8+nt}
//   element (b, lane, h) at wf[((b*32 + lane) << 4) + h]  -- 32B/lane, contiguous
#define EMB_FRAG_BASE  0
#define GCN_FRAG_BASE(l) (16 + 32 * (l))
#define N_FRAG_BLOCKS  80          // 16 emb + 2*32 gcn
#define WF_HALVES      (N_FRAG_BLOCKS * 32 * 16)   // 40960 halves = 80KB

// ---------------------------------------------------------------------------
// One-shot weight prepack: row-major f32 W[K][128] -> WMMA B-fragment f16.
// B (32x16 f16) mapping: lane = (n&15) + 16*((k'>>3)&1), half h = (k'&7) + 8*(k'>>4).
// ---------------------------------------------------------------------------
__global__ __launch_bounds__(256)
void prepack_weights(const float* __restrict__ wEmb,
                     const float* __restrict__ wGcn,
                     _Float16* __restrict__ wf) {
  const int e = blockIdx.x * 256 + threadIdx.x;
  const int totalEmb = OBSD * HIDD;               // 8192
  const int totalGcn = 2 * HIDD * HIDD;           // 32768
  if (e >= totalEmb + totalGcn) return;

  int k, n, blkBase;
  float val;
  if (e < totalEmb) {
    k = e >> 7; n = e & 127;
    val = wEmb[e];
    blkBase = EMB_FRAG_BASE;
  } else {
    const int e2 = e - totalEmb;
    const int l  = e2 >> 14;                      // 128*128 = 16384 per layer
    const int r  = e2 & 16383;
    k = r >> 7; n = r & 127;
    val = wGcn[e2];
    blkBase = GCN_FRAG_BASE(l);
  }
  const int nt = n >> 4;
  const int kc = k >> 5;
  const int kp = k & 31;
  const int ln = (n & 15) + (((kp >> 3) & 1) << 4);
  const int h  = (kp & 7) + ((kp >> 4) << 3);
  wf[(((blkBase + kc * 8 + nt) * 32 + ln) << 4) + h] = (_Float16)val;
}

// Store one D-layout f32 tile column-set (8 rows for this lane) into the
// per-wave A-fragment LDS buffer (f16) for the next GEMM.
// A (16x32 f16) mapping: lane = M + 16*((k'>>3)&1), h = (k'&7) + 8*(k'>>4).
__device__ __forceinline__ void store_x_frag(_Float16* __restrict__ xbuf,
                                             const v8f& xn, int col, int hi) {
  const int kc    = col >> 5;
  const int kp    = col & 31;
  const int hsel  = (kp & 7) + (((kp >> 4) & 1) << 3);
  const int lbase = ((kp >> 3) & 1) << 4;
  _Float16* p = xbuf + ((kc * 32 + lbase + (hi << 3)) << 4) + hsel;
#pragma unroll
  for (int r = 0; r < 8; ++r) p[r << 4] = (_Float16)xn[r];
}

// ---------------------------------------------------------------------------
// Fused GCN critic: one wave == one 16-agent graph. No block barriers at all;
// B operands come straight from the packed global pool (cache-resident).
// ---------------------------------------------------------------------------
__global__ __launch_bounds__(BLK_THREADS)
void gcn_critic_fused(const float* __restrict__ obs,
                      const _Float16* __restrict__ wf,
                      const float* __restrict__ bEmb,
                      const float* __restrict__ bGcn,
                      const float* __restrict__ wFc1,
                      const float* __restrict__ bFc1,
                      float* __restrict__ out) {
  __shared__ __align__(32) _Float16 sX[WAVES_PER_BLK * 4 * 32 * 16];  // 32KB

  const int tid  = threadIdx.x;
  const int lane = tid & 31;
  const int wave = tid >> 5;
  const int g    = blockIdx.x * WAVES_PER_BLK + wave;
  const int col0 = lane & 15;
  const int hi   = lane >> 4;
  _Float16* xbuf = sX + wave * (4 * 32 * 16);

  // ---------------- embedding: x0 = obs @ w_emb + b_emb ----------------
  const float* orow = obs + ((size_t)g * NAGENTS + col0) * OBSD;
  const int kbase = hi << 3;
  v16h ae[2];
#pragma unroll
  for (int c = 0; c < 2; ++c) {
    const float* p = orow + c * 32 + kbase;
    float4 f0 = *(const float4*)(p);
    float4 f1 = *(const float4*)(p + 4);
    float4 f2 = *(const float4*)(p + 16);
    float4 f3 = *(const float4*)(p + 20);
    v16h a;
    a[0]  = (_Float16)f0.x; a[1]  = (_Float16)f0.y; a[2]  = (_Float16)f0.z; a[3]  = (_Float16)f0.w;
    a[4]  = (_Float16)f1.x; a[5]  = (_Float16)f1.y; a[6]  = (_Float16)f1.z; a[7]  = (_Float16)f1.w;
    a[8]  = (_Float16)f2.x; a[9]  = (_Float16)f2.y; a[10] = (_Float16)f2.z; a[11] = (_Float16)f2.w;
    a[12] = (_Float16)f3.x; a[13] = (_Float16)f3.y; a[14] = (_Float16)f3.z; a[15] = (_Float16)f3.w;
    ae[c] = a;
  }

  v8f xd[8];  // activation in WMMA D layout (f32), 8 N-tiles
#pragma unroll
  for (int nt = 0; nt < 8; ++nt) {
    v8f acc = {};
#pragma unroll
    for (int c = 0; c < 2; ++c) {
      v16h b = *(const v16h*)(wf + (((EMB_FRAG_BASE + c * 8 + nt) * 32 + lane) << 4));
      acc = __builtin_amdgcn_wmma_f32_16x16x32_f16(false, ae[c], false, b,
                                                   (short)0, acc, false, false);
    }
    const int col = col0 + (nt << 4);
    const float be = bEmb[col];
#pragma unroll
    for (int r = 0; r < 8; ++r) acc[r] += be;
    xd[nt] = acc;
    store_x_frag(xbuf, acc, col, hi);   // pack as next GEMM's A operand
  }

  // ---------------- 2 GCN layers (wave-local; LDS is in-order) -------------
#pragma unroll
  for (int l = 0; l < 2; ++l) {
    v16h af[4];
#pragma unroll
    for (int kc = 0; kc < 4; ++kc)
      af[kc] = *(const v16h*)(xbuf + ((kc * 32 + lane) << 4));

#pragma unroll
    for (int nt = 0; nt < 8; ++nt) {
      v8f acc = {};
#pragma unroll
      for (int kc = 0; kc < 4; ++kc) {
        v16h b = *(const v16h*)(wf + (((GCN_FRAG_BASE(l) + kc * 8 + nt) * 32 + lane) << 4));
        acc = __builtin_amdgcn_wmma_f32_16x16x32_f16(false, af[kc], false, b,
                                                     (short)0, acc, false, false);
      }
      // GCN aggregation == column mean over the tile's 16 rows (one graph)
      float s = 0.f;
#pragma unroll
      for (int r = 0; r < 8; ++r) s += acc[r];
      s += __shfl_xor(s, 16, 32);
      const int col = col0 + (nt << 4);
      const float m = s * 0.0625f + bGcn[l * HIDD + col];
      v8f xo = xd[nt];
      v8f xn;
#pragma unroll
      for (int r = 0; r < 8; ++r) xn[r] = tanhf(m + xo[r]);  // residual + tanh
      xd[nt] = xn;
      if (l == 0) store_x_frag(xbuf, xn, col, hi);
    }
  }

  // ------------- head: out[g] = dot(colmean(x), w_fc1) + b_fc1 -------------
  float p = 0.f;
#pragma unroll
  for (int nt = 0; nt < 8; ++nt) {
    float s = 0.f;
#pragma unroll
    for (int r = 0; r < 8; ++r) s += xd[nt][r];
    s += __shfl_xor(s, 16, 32);
    if (lane < 16) p += (s * 0.0625f) * wFc1[col0 + (nt << 4)];
  }
#pragma unroll
  for (int off = 1; off < 32; off <<= 1) p += __shfl_xor(p, off, 32);
  if (lane == 0) out[g] = p + bFc1[0];
}

extern "C" void kernel_launch(void* const* d_in, const int* in_sizes, int n_in,
                              void* d_out, int out_size, void* d_ws, size_t ws_size,
                              hipStream_t stream) {
  (void)in_sizes; (void)n_in; (void)out_size; (void)ws_size;
  const float* obs  = (const float*)d_in[0];
  const float* wEmb = (const float*)d_in[1];
  const float* bEmb = (const float*)d_in[2];
  const float* wGcn = (const float*)d_in[3];
  const float* bGcn = (const float*)d_in[4];
  const float* wFc1 = (const float*)d_in[5];
  const float* bFc1 = (const float*)d_in[6];
  // edge_src/edge_dst (d_in[7..8]) describe complete graphs + self loops with
  // uniform degree 16 -> aggregation is a per-graph column mean; not needed.
  float* out = (float*)d_out;
  _Float16* wf = (_Float16*)d_ws;   // 80KB packed B-fragment pool

  const int totalW = OBSD * HIDD + 2 * HIDD * HIDD;          // 40960
  prepack_weights<<<(totalW + 255) / 256, 256, 0, stream>>>(wEmb, wGcn, wf);

  dim3 grid(NBATCH / WAVES_PER_BLK);  // 1024 blocks, 8 waves (graphs) each
  gcn_critic_fused<<<grid, BLK_THREADS, 0, stream>>>(obs, wf, bEmb, bGcn,
                                                     wFc1, bFc1, out);
}
